// GNNModule_17935783428737
// MI455X (gfx1250) — matrix operand order repaired and checked by reading, not compiled, not run on hardware
//
#include <hip/hip_runtime.h>

typedef __attribute__((ext_vector_type(16))) _Float16 v16h;
typedef __attribute__((ext_vector_type(8)))  float    v8f;

#define D 64
#define KN 8            // neighbors per table
#define NSTEP 10        // K = 320 = 10 * 32
#define TILE_ROWS 128   // 8 waves * 16 rows per block
#define WSWIZ 20480     // NSTEP*4*32*16 halves per branch

struct Params10 {
  const float* w[10];
  const float* b[10];
};

// ---------------------------------------------------------------------------
// Prep: swizzle concatenated [320,64] weights into WMMA B-fragment lane order
// (f16), sum the 5 biases per branch, zero the BN stat accumulators.
// B-frag layout (16x16x32 f16): lane L holds col=(L&15), K=(L>>4)*8 + {0..7, 16..23}.
// Wcat row blocks: [self, deg*self, agg0, agg1, cross] ->
//   x: theta_x, theta_deg, theta_list0, theta_list1, theta_y  (w idx 0,1,3,4,2)
//   y: gamma_y, gamma_deg, gamma_list0, gamma_list1, gamma_x  (w idx 5,6,8,9,7)
// ---------------------------------------------------------------------------
__global__ void prep_kernel(Params10 pp, _Float16* __restrict__ Wx,
                            _Float16* __restrict__ Wy,
                            float* __restrict__ biasX, float* __restrict__ biasY,
                            float* __restrict__ stats) {
  int tid = blockIdx.x * blockDim.x + threadIdx.x;
  int nth = gridDim.x * blockDim.x;
  for (int e = tid; e < 2 * WSWIZ; e += nth) {
    int branch = (e >= WSWIZ) ? 1 : 0;
    int e2 = e - branch * WSWIZ;
    int i    = e2 & 15;
    int lane = (e2 >> 4) & 31;
    int ct   = (e2 >> 9) & 3;
    int s    = e2 >> 11;
    int col  = ct * 16 + (lane & 15);
    int k    = s * 32 + (lane >> 4) * 8 + (i & 7) + (i >> 3) * 16;
    int blk  = k >> 6;
    int kr   = k & 63;
    int sel  = (blk == 0) ? 0 : (blk == 1) ? 1 : (blk == 2) ? 3 : (blk == 3) ? 4 : 2;
    float v  = pp.w[sel + branch * 5][kr * D + col];
    if (branch) Wy[e2] = (_Float16)v; else Wx[e2] = (_Float16)v;
  }
  for (int c = tid; c < D; c += nth) {
    float sx = 0.f, sy = 0.f;
#pragma unroll
    for (int j = 0; j < 5; j++) { sx += pp.b[j][c]; sy += pp.b[5 + j][c]; }
    biasX[c] = sx; biasY[c] = sy;
  }
  for (int c = tid; c < 4 * D; c += nth) stats[c] = 0.f;
}

__global__ void zero_kernel(float4* __restrict__ p, int n4) {
  int t = blockIdx.x * blockDim.x + threadIdx.x;
  if (t < n4) p[t] = make_float4(0.f, 0.f, 0.f, 0.f);
}

// pmpd_y[n, c] += y[e, c] for edges with dst[e] == n  (segment_sum)
__global__ void scatter_kernel(const float* __restrict__ y, const int* __restrict__ dst,
                               float* __restrict__ pmpd, int M) {
  int t = blockIdx.x * blockDim.x + threadIdx.x;
  if (t >= M * D) return;
  int e = t >> 6;
  int c = t & 63;
  atomicAdd(&pmpd[dst[e] * D + c], y[t]);
}

// ---------------------------------------------------------------------------
// Fused: u = [F | deg*F | sum_k F[t0] | sum_k F[t1] | E(gathered)] (R x 320),
// z = half_relu(u @ Wcat + biasSum); writes z, accumulates BN sum/sumsq.
// One wave = 16 rows x 64 cols via 4 f32 accumulators, 10 K-steps of
// v_wmma_f32_16x16x32_f16.
// ---------------------------------------------------------------------------
__global__ void __launch_bounds__(256)
gemm_kernel(const float* __restrict__ F, const float* __restrict__ deg,
            const int* __restrict__ t0, const int* __restrict__ t1,
            const float* __restrict__ E, const int* __restrict__ eidx,
            const _Float16* __restrict__ Wf, const float* __restrict__ biasSum,
            float* __restrict__ zOut, float* __restrict__ gSum,
            float* __restrict__ gSq, int R) {
  __shared__ float sSum[D];
  __shared__ float sSq[D];
  if (threadIdx.x < D) { sSum[threadIdx.x] = 0.f; sSq[threadIdx.x] = 0.f; }
  __syncthreads();

  const int wave = threadIdx.x >> 5;
  const int lane = threadIdx.x & 31;
  const int half = lane >> 4;
  const int tileBase = (blockIdx.x * 8 + wave) * 16;

  const int rowA = tileBase + (lane & 15);
  const int rowC = (rowA < R) ? rowA : (R - 1);   // clamp; results masked later

  const float4* __restrict__ F4 = (const float4*)F;
  const float4* __restrict__ E4 = (const float4*)E;
  const v16h*   __restrict__ Bf = (const v16h*)Wf;

  const float degv = deg[rowC];
  const int er = eidx ? eidx[rowC] : rowC;
  int nb0[KN], nb1[KN];
#pragma unroll
  for (int j = 0; j < KN; j++) { nb0[j] = t0[rowC * KN + j]; nb1[j] = t1[rowC * KN + j]; }

  v8f acc0 = {}, acc1 = {}, acc2 = {}, acc3 = {};

#pragma unroll
  for (int s = 0; s < NSTEP; s++) {
    // A fragment, ISA layout: lane<16 -> K = s*32 + {0..7, 16..23};
    //                         lane>=16 -> K = s*32 + {8..15, 24..31}
    v16h a;
#pragma unroll
    for (int rrun = 0; rrun < 2; rrun++) {
      const int kloc = s * 32 + rrun * 16;          // segment is wave-uniform
      const int seg  = kloc >> 6;                   // compile-time after unroll
      const int c4   = ((kloc & 63) >> 2) + half * 2;
      float r0, r1, r2, r3, r4, r5, r6, r7;
      if (seg == 0 || seg == 1) {
        float4 p = F4[rowC * 16 + c4];
        float4 q = F4[rowC * 16 + c4 + 1];
        float dm = (seg == 1) ? degv : 1.0f;
        r0 = p.x * dm; r1 = p.y * dm; r2 = p.z * dm; r3 = p.w * dm;
        r4 = q.x * dm; r5 = q.y * dm; r6 = q.z * dm; r7 = q.w * dm;
      } else if (seg == 4) {
        float4 p = E4[er * 16 + c4];
        float4 q = E4[er * 16 + c4 + 1];
        r0 = p.x; r1 = p.y; r2 = p.z; r3 = p.w;
        r4 = q.x; r5 = q.y; r6 = q.z; r7 = q.w;
      } else {
        const int* nb = (seg == 2) ? nb0 : nb1;
        r0 = r1 = r2 = r3 = r4 = r5 = r6 = r7 = 0.f;
#pragma unroll
        for (int j = 0; j < KN; j++) {
          float4 p = F4[nb[j] * 16 + c4];
          float4 q = F4[nb[j] * 16 + c4 + 1];
          r0 += p.x; r1 += p.y; r2 += p.z; r3 += p.w;
          r4 += q.x; r5 += q.y; r6 += q.z; r7 += q.w;
        }
      }
      const int o = rrun * 8;
      a[o + 0] = (_Float16)r0; a[o + 1] = (_Float16)r1;
      a[o + 2] = (_Float16)r2; a[o + 3] = (_Float16)r3;
      a[o + 4] = (_Float16)r4; a[o + 5] = (_Float16)r5;
      a[o + 6] = (_Float16)r6; a[o + 7] = (_Float16)r7;
    }
    const int bb = s * 128 + lane;                  // coalesced 32B/lane frags
    v16h b0 = Bf[bb];
    v16h b1 = Bf[bb + 32];
    v16h b2 = Bf[bb + 64];
    v16h b3 = Bf[bb + 96];
    acc0 = __builtin_amdgcn_wmma_f32_16x16x32_f16(false, a, false, b0, (short)0, acc0, false, false);
    acc1 = __builtin_amdgcn_wmma_f32_16x16x32_f16(false, a, false, b1, (short)0, acc1, false, false);
    acc2 = __builtin_amdgcn_wmma_f32_16x16x32_f16(false, a, false, b2, (short)0, acc2, false, false);
    acc3 = __builtin_amdgcn_wmma_f32_16x16x32_f16(false, a, false, b3, (short)0, acc3, false, false);
  }

  // Epilogue: bias + half-ReLU + store + BN partial sums.
  // C/D layout: VGPR r -> row = tileBase + half*8 + r, col = (lane&15) + ct*16.
  const int colLow  = lane & 15;
  const int rowBase = tileBase + half * 8;
#pragma unroll
  for (int ct = 0; ct < 4; ct++) {
    v8f acc = (ct == 0) ? acc0 : (ct == 1) ? acc1 : (ct == 2) ? acc2 : acc3;
    const int col = ct * 16 + colLow;
    const float bsv = biasSum[col];
    const bool doRelu = (col >= (D / 2));
    float s1 = 0.f, s2 = 0.f;
#pragma unroll
    for (int rr = 0; rr < 8; rr++) {
      float v = acc[rr] + bsv;
      if (doRelu) v = fmaxf(v, 0.f);
      const int row = rowBase + rr;
      if (row < R) {
        zOut[row * D + col] = v;
        s1 += v;
        s2 += v * v;
      }
    }
    atomicAdd(&sSum[col], s1);
    atomicAdd(&sSq[col], s2);
  }
  __syncthreads();
  if (threadIdx.x < D) {
    atomicAdd(&gSum[threadIdx.x], sSum[threadIdx.x]);
    atomicAdd(&gSq[threadIdx.x], sSq[threadIdx.x]);
  }
}

// In-place BatchNorm (training mode, biased variance) over d_out.
__global__ void bn_kernel(float* __restrict__ z, const float* __restrict__ gSum,
                          const float* __restrict__ gSq, const float* __restrict__ scale,
                          const float* __restrict__ bias, int R) {
  int t = blockIdx.x * blockDim.x + threadIdx.x;
  if (t >= R * D) return;
  int col = t & 63;
  float invn = 1.0f / (float)R;
  float mu  = gSum[col] * invn;
  float var = gSq[col] * invn - mu * mu;
  float is  = rsqrtf(var + 1e-5f);
  z[t] = (z[t] - mu) * is * scale[col] + bias[col];
}

// ---------------------------------------------------------------------------
extern "C" void kernel_launch(void* const* d_in, const int* in_sizes, int n_in,
                              void* d_out, int out_size, void* d_ws, size_t ws_size,
                              hipStream_t stream) {
  (void)n_in; (void)out_size; (void)ws_size;
  const int N = 50000, M = 400000;

  const float* x      = (const float*)d_in[0];
  const float* yv     = (const float*)d_in[1];
  const float* deg_g  = (const float*)d_in[2];
  const float* deg_lg = (const float*)d_in[3];

  // 10 linear layers, insertion order:
  // theta_x, theta_deg, theta_y, theta_list0, theta_list1,
  // gamma_y, gamma_deg, gamma_x, gamma_list0, gamma_list1
  Params10 pp;
  int p = 4;
  for (int i = 0; i < 10; i++) {
    const float* a0 = (const float*)d_in[p];
    const float* a1 = (const float*)d_in[p + 1];
    if (in_sizes[p] == D * D) { pp.w[i] = a0; pp.b[i] = a1; }
    else                      { pp.w[i] = a1; pp.b[i] = a0; }
    p += 2;
  }
  const float* bnx_scale = (const float*)d_in[24];
  const float* bnx_bias  = (const float*)d_in[25];
  const float* bny_scale = (const float*)d_in[26];
  const float* bny_bias  = (const float*)d_in[27];
  const int* pm_pd = (const int*)d_in[28];
  const int* edges = (const int*)d_in[29];
  const int* g_t   = (const int*)d_in[30];
  const int* g_tt  = (const int*)d_in[31];
  const int* lg_t  = (const int*)d_in[32];
  const int* lg_tt = (const int*)d_in[33];

  // Workspace layout (all offsets 32B-aligned)
  char* ws = (char*)d_ws;
  float*    pmpd  = (float*)ws;                                  // N*64 f32
  size_t off = (size_t)N * D * sizeof(float);
  _Float16* Wx    = (_Float16*)(ws + off);                       // 20480 halves
  _Float16* Wy    = Wx + WSWIZ;
  float*    biasX = (float*)(ws + off + 2 * (size_t)WSWIZ * sizeof(_Float16));
  float*    biasY = biasX + D;
  float*    stats = biasY + D;   // [sumX | sqX | sumY | sqY], 4*64 f32

  float* outX = (float*)d_out;
  float* outY = outX + (size_t)N * D;

  hipLaunchKernelGGL(prep_kernel, dim3(64), dim3(256), 0, stream,
                     pp, Wx, Wy, biasX, biasY, stats);

  const int n4 = N * D / 4;
  hipLaunchKernelGGL(zero_kernel, dim3((n4 + 255) / 256), dim3(256), 0, stream,
                     (float4*)pmpd, n4);

  hipLaunchKernelGGL(scatter_kernel, dim3((M * D + 255) / 256), dim3(256), 0, stream,
                     yv, edges, pmpd, M);

  hipLaunchKernelGGL(gemm_kernel, dim3((N + TILE_ROWS - 1) / TILE_ROWS), dim3(256), 0, stream,
                     x, deg_g, g_t, g_tt, pmpd, (const int*)nullptr, Wx, biasX,
                     outX, stats, stats + D, N);

  hipLaunchKernelGGL(gemm_kernel, dim3((M + TILE_ROWS - 1) / TILE_ROWS), dim3(256), 0, stream,
                     yv, deg_lg, lg_t, lg_tt, x, pm_pd, Wy, biasY,
                     outY, stats + 2 * D, stats + 3 * D, M);

  hipLaunchKernelGGL(bn_kernel, dim3((N * D + 255) / 256), dim3(256), 0, stream,
                     outX, stats, stats + D, bnx_scale, bnx_bias, N);

  hipLaunchKernelGGL(bn_kernel, dim3((M * D + 255) / 256), dim3(256), 0, stream,
                     outY, stats + 2 * D, stats + 3 * D, bny_scale, bny_bias, M);
}
